// AMPBlock2_65377992180284
// MI455X (gfx1250) — matrix operand order repaired and checked by reading, not compiled, not run on hardware
//
#include <hip/hip_runtime.h>
#include <math.h>

#define BATCH 8
#define CH    512
#define TLEN  8192
#define FK    12

typedef __attribute__((ext_vector_type(16))) __bf16 v16bf;
typedef __attribute__((ext_vector_type(8)))  float  v8f;

static __device__ __forceinline__ unsigned short f2bf(float f) {
  unsigned u = __float_as_uint(f);
  unsigned r = u + 0x7fffu + ((u >> 16) & 1u);
  return (unsigned short)(r >> 16);
}

static __device__ __forceinline__ float bessel_i0(float x) {
  float t = 1.f, s = 1.f;
  float hx = 0.5f * x;
  #pragma unroll
  for (int k = 1; k <= 24; ++k) {
    float q = hx / (float)k;
    t *= q * q;
    s += t;
  }
  return s;
}

// ---------------------------------------------------------------------------
// Pack weights (C,C,3) fp32 -> bf16 WMMA A-fragment layout.
// Fragment f = (k*32 + mtile)*16 + ci_chunk ; 32 lanes x 16 bf16 each.
// Lane l (row m = mtile*16 + l%16):
//   l < 16 : K order = {0..7, 16..23}
//   l >= 16: K order = {8..15, 24..31}
// ---------------------------------------------------------------------------
__global__ __launch_bounds__(32) void pack_kernel(const float* __restrict__ w,
                                                  unsigned short* __restrict__ pw) {
  const int f  = blockIdx.x;           // (k*32 + mt)*16 + cc
  const int cc = f & 15;
  const int mt = (f >> 4) & 31;
  const int k  = f >> 9;
  const int l  = threadIdx.x;
  const int m  = mt * 16 + (l & 15);
  const int kb = (l < 16) ? 0 : 8;
  unsigned short* dst = pw + (size_t)f * 512 + (size_t)l * 16;
  #pragma unroll
  for (int v = 0; v < 16; ++v) {
    int kk = (v < 8 ? v : v + 8) + kb;       // 0..31 within chunk
    int ci = cc * 32 + kk;
    dst[v] = f2bf(w[((size_t)m * CH + ci) * 3 + k]);
  }
}

// ---------------------------------------------------------------------------
// Fused anti-aliased snake activation:
//   edge-pad -> 2x polyphase upsample (12-tap kaiser-sinc) -> snake
//   -> 12-tap downsample (stride 2) -> bf16 output.
// One block handles 512 output samples of one (b,c) row.
// ---------------------------------------------------------------------------
__global__ __launch_bounds__(256) void act_kernel(const float* __restrict__ x,
                                                  const float* __restrict__ alpha_log,
                                                  unsigned short* __restrict__ xt) {
  __shared__ float s_filt[FK];
  __shared__ float s_y[2 * 512 + FK];   // 1036 upsampled+snaked samples
  const int row = blockIdx.x;           // b*C + c
  const int c   = row & (CH - 1);
  const int t0  = blockIdx.y * 512;
  const int tid = threadIdx.x;

  if (tid == 0) {
    const float pi   = 3.14159265358979f;
    const float A    = 2.285f * (FK / 2 - 1) * pi * (4.f * 0.3f) + 7.95f;
    const float beta = 0.1102f * (A - 8.7f);
    const float ib   = bessel_i0(beta);
    float tmp[FK];
    float ssum = 0.f;
    #pragma unroll
    for (int n = 0; n < FK; ++n) {
      float rr  = 2.f * n / (FK - 1) - 1.f;
      float wk  = bessel_i0(beta * sqrtf(fmaxf(0.f, 1.f - rr * rr))) / ib;
      float tm  = (float)n - (FK / 2) + 0.5f;
      float xs  = 0.5f * tm;                 // 2*cutoff*time, cutoff = 0.25
      float snc = (fabsf(xs) < 1e-6f) ? 1.f : __sinf(pi * xs) / (pi * xs);
      tmp[n] = 0.5f * wk * snc;
      ssum  += tmp[n];
    }
    #pragma unroll
    for (int n = 0; n < FK; ++n) s_filt[n] = tmp[n] / ssum;
  }
  __syncthreads();

  const float a     = __expf(alpha_log[c]);
  const float inv_a = 1.f / (a + 1e-9f);
  const float* xr   = x + (size_t)row * TLEN;

  // upsampled coords u in [2*t0 - 5, 2*t0 + 1030]
  const int ubase = 2 * t0 - 5;
  for (int i = tid; i < 2 * 512 + FK; i += 256) {
    int u = ubase + i;
    u = min(max(u, 0), 2 * TLEN - 1);
    const int p = u & 1;
    const int m = u >> 1;
    float acc = 0.f;
    #pragma unroll
    for (int q = 0; q < 6; ++q) {
      int mi = min(max(m - 2 + q, 0), TLEN - 1);      // edge pad
      acc += s_filt[11 - (2 * q + (p ^ 1))] * xr[mi]; // reversed-filter phase
    }
    float y  = 2.f * acc;
    float sy = __sinf(a * y);
    s_y[i] = y + sy * sy * inv_a;                      // snake
  }
  __syncthreads();

  unsigned short* xo = xt + (size_t)row * TLEN + t0;
  #pragma unroll
  for (int rep = 0; rep < 2; ++rep) {
    const int tl = rep * 256 + tid;
    float acc = 0.f;
    #pragma unroll
    for (int j = 0; j < FK; ++j) acc += s_filt[j] * s_y[2 * tl + j];
    xo[tl] = f2bf(acc);
  }
}

// ---------------------------------------------------------------------------
// Implicit-GEMM dilated conv via v_wmma_f32_16x16x32_bf16.
// Block tile: M=128 (co) x N=64 (t); 8 waves, each wave: 16(M) x 64(N).
// K loop: 16 chunks of 32 channels x 3 taps.
// xt staged transposed into double-buffered LDS [t][ci-pair] with a
// 20-dword row stride: every B fragment base is 16B-aligned (2 x
// ds_load_b128 per fragment) and rows are bank-conflict-free.
// Staging is fully branchless: addresses are clamped + cndmask-zeroed, and
// buffers are padded to 80 rows so every thread stages exactly 5 rows.
// Next chunk's global loads issue before the current chunk's 12 WMMAs.
// ---------------------------------------------------------------------------
struct AFrag { uint4 q[2]; };
struct BFrag { uint4 q[2]; };

// branchless per-thread staging load: 4 full rows + 1 padded tail row
template <int DIL>
static __device__ __forceinline__ void ld_chunk(
    const unsigned short* __restrict__ s0, const unsigned short* __restrict__ s1,
    int t0, int jr, unsigned (&r)[5]) {
  #pragma unroll
  for (int it = 0; it < 5; ++it) {
    const int rr  = (it < 4) ? (jr + 16 * it) : (64 + jr);
    const int tg  = t0 - DIL + rr;
    const int tgc = min(max(tg, 0), TLEN - 1);
    unsigned v = (unsigned)s0[tgc] | ((unsigned)s1[tgc] << 16);
    if (tg != tgc) v = 0;                   // v_cndmask, no branch
    r[it] = v;
  }
}

static __device__ __forceinline__ void st_chunk(unsigned* buf, int cl2, int jr,
                                                const unsigned (&r)[5]) {
  #pragma unroll
  for (int it = 0; it < 5; ++it) {
    const int rr = (it < 4) ? (jr + 16 * it) : (64 + jr);  // rows >= tile: pad
    buf[rr * 20 + cl2] = r[it];
  }
}

template <int DIL>
__global__ __launch_bounds__(256) void gemm_kernel(
    const unsigned short* __restrict__ xt, const unsigned short* __restrict__ pw,
    const float* __restrict__ bias, const float* res, float* out) {
  __shared__ __align__(16) unsigned s_x[2][80 * 20];  // 80 rows: tile + pad

  const int tid  = threadIdx.x;
  const int lane = tid & 31;
  const int wv   = tid >> 5;             // 0..7 -> M sub-tile
  const int t0   = blockIdx.x * 64;
  const int cob  = blockIdx.y;           // 0..3
  const int b    = blockIdx.z;
  const int co_base = cob * 128 + wv * 16;

  v8f zero = {};
  v8f acc[4];
  #pragma unroll
  for (int nt = 0; nt < 4; ++nt) acc[nt] = zero;

  const unsigned short* xb = xt + (size_t)b * CH * TLEN;
  const int cl2 = tid & 15;              // ci pair within chunk
  const int jr  = tid >> 4;              // row phase for staging (0..15)

  const int khalf = (lane < 16) ? 0 : 4; // dword offset: K half per lane
  const int ncol  = lane & 15;           // N within 16-wide tile
  // wave-invariant weight base: fragments for this (cob, wv), tap-major
  const unsigned short* pw_wave = pw + ((size_t)(cob * 8 + wv) * 16) * 512
                                     + (size_t)lane * 16;

  // ---- prologue: stage chunk 0 ----
  {
    unsigned regs[5];
    const unsigned short* s0 = xb + (size_t)(2 * cl2) * TLEN;
    ld_chunk<DIL>(s0, s0 + TLEN, t0, jr, regs);
    st_chunk(&s_x[0][0], cl2, jr, regs);
  }
  __syncthreads();

  for (int cc = 0; cc < 16; ++cc) {
    const int cur = cc & 1;

    // issue next chunk's global loads early (overlaps with WMMAs below)
    unsigned nregs[5];
    if (cc + 1 < 16) {
      const unsigned short* s0 = xb + (size_t)((cc + 1) * 32 + 2 * cl2) * TLEN;
      ld_chunk<DIL>(s0, s0 + TLEN, t0, jr, nregs);
      if (cc + 2 < 16) {  // hint chunk cc+2 into cache (global_prefetch_b8)
        __builtin_prefetch(xb + (size_t)((cc + 2) * 32 + 2 * cl2) * TLEN + t0 + jr * 4,
                           0, 1);
      }
    }

    // all 3 tap A fragments: one clause of 6 global_load_b128
    AFrag af[3];
    #pragma unroll
    for (int k = 0; k < 3; ++k) {
      const uint4* ap = (const uint4*)(pw_wave + ((size_t)k * 32 * 16 + cc) * 512);
      af[k].q[0] = ap[0];
      af[k].q[1] = ap[1];
    }

    const unsigned* s_base = &s_x[cur][0] + ncol * 20 + khalf;  // per-lane base
    #pragma unroll
    for (int k = 0; k < 3; ++k) {
      // 4 distinct B fragments (2 x ds_load_b128 each), then 4 WMMAs
      BFrag bf[4];
      #pragma unroll
      for (int nt = 0; nt < 4; ++nt) {
        const uint4* bp = (const uint4*)(s_base + (nt * 16 + k * DIL) * 20);
        bf[nt].q[0] = bp[0];
        bf[nt].q[1] = bp[2];
      }
      const v16bf av = *(const v16bf*)&af[k];
      #pragma unroll
      for (int nt = 0; nt < 4; ++nt) {
        acc[nt] = __builtin_amdgcn_wmma_f32_16x16x32_bf16(
            false, av, false, *(const v16bf*)&bf[nt], (short)0, acc[nt],
            false, false);
      }
    }

    // write next chunk into the other buffer (readers of it synced below)
    if (cc + 1 < 16) st_chunk(&s_x[1 - cur][0], cl2, jr, nregs);
    __syncthreads();
  }

  // epilogue: bias + residual, fp32 out
  const int mo = (lane < 16) ? 0 : 8;
  float bv[8];
  #pragma unroll
  for (int r = 0; r < 8; ++r) bv[r] = bias[co_base + mo + r];
  #pragma unroll
  for (int nt = 0; nt < 4; ++nt) {
    const int tg = t0 + nt * 16 + ncol;
    #pragma unroll
    for (int r = 0; r < 8; ++r) {
      const size_t oidx = ((size_t)b * CH + co_base + mo + r) * TLEN + tg;
      out[oidx] = acc[nt][r] + bv[r] + res[oidx];
    }
  }
}

// ---------------------------------------------------------------------------
extern "C" void kernel_launch(void* const* d_in, const int* in_sizes, int n_in,
                              void* d_out, int out_size, void* d_ws, size_t ws_size,
                              hipStream_t stream) {
  (void)in_sizes; (void)n_in; (void)out_size; (void)ws_size;
  const float* x  = (const float*)d_in[0];
  const float* w1 = (const float*)d_in[1];
  const float* b1 = (const float*)d_in[2];
  const float* a1 = (const float*)d_in[3];
  const float* w2 = (const float*)d_in[4];
  const float* b2 = (const float*)d_in[5];
  const float* a2 = (const float*)d_in[6];
  float* outp = (float*)d_out;

  unsigned short* wsX = (unsigned short*)d_ws;                       // 64 MiB bf16 activations
  unsigned short* wsW = (unsigned short*)((char*)d_ws +
                         (size_t)BATCH * CH * TLEN * sizeof(unsigned short)); // 1.5 MiB packed W

  dim3 actG(BATCH * CH, TLEN / 512), actB(256);
  dim3 gmG(TLEN / 64, CH / 128, BATCH), gmB(256);

  // stage 1 (dilation 1): residual from x, write x1 to d_out
  pack_kernel<<<1536, 32, 0, stream>>>(w1, wsW);
  act_kernel<<<actG, actB, 0, stream>>>(x, a1, wsX);
  gemm_kernel<1><<<gmG, gmB, 0, stream>>>(wsX, wsW, b1, x, outp);

  // stage 2 (dilation 3): residual from x1 (in-place on d_out)
  pack_kernel<<<1536, 32, 0, stream>>>(w2, wsW);
  act_kernel<<<actG, actB, 0, stream>>>(outp, a2, wsX);
  gemm_kernel<3><<<gmG, gmB, 0, stream>>>(wsX, wsW, b2, outp, outp);
}
